// Block_27161373180410
// MI455X (gfx1250) — matrix operand order repaired and checked by reading, not compiled, not run on hardware
//
#include <hip/hip_runtime.h>
#include <stdint.h>

// ---------------------------------------------------------------------------
// CDNA5 (gfx1250) transformer block: RMSNorm -> QKV -> causal flash attention
// -> proj(+res) -> RMSNorm -> GLU FFN(+res).  All GEMMs via
// v_wmma_f32_16x16x32_bf16 (wave32).  GEMM tiles staged with a double-
// buffered global_load_async_to_lds_b128 pipeline (ASYNCcnt).
// ---------------------------------------------------------------------------

typedef __attribute__((ext_vector_type(16))) __bf16 v16bf;
typedef __attribute__((ext_vector_type(8)))  float  v8f;

static constexpr int B_ = 2, T_ = 2048, D_ = 1024, H_ = 16, HD_ = 64;
static constexpr int M_ = B_ * T_;          // 4096 rows of activations
static constexpr float EPS_ = 1e-6f;

union Frag16 { v16bf v; uint4 q[2]; };

__device__ __forceinline__ unsigned short f2bf(float f) {
  unsigned int u = __builtin_bit_cast(unsigned int, f);
  u += 0x7fffu + ((u >> 16) & 1u);          // round-to-nearest-even
  return (unsigned short)(u >> 16);
}
__device__ __forceinline__ float bf2f(unsigned short h) {
  unsigned int u = ((unsigned int)h) << 16;
  return __builtin_bit_cast(float, u);
}

__device__ __forceinline__ v8f wmma_bf16(v16bf a, v16bf b, v8f c) {
  return __builtin_amdgcn_wmma_f32_16x16x32_bf16(
      false, a, false, b, (short)0, c, false, false);
}

// Async global->LDS copy, 16B per lane.  lds_off = 32-bit group-segment
// offset (flat LDS aperture keeps the offset in addr[31:0]).
__device__ __forceinline__ void async_load_b128(unsigned lds_off,
                                                const void* gaddr) {
  asm volatile("global_load_async_to_lds_b128 %0, %1, off"
               :: "v"(lds_off), "v"(gaddr) : "memory");
}
// Wait until at most the 6 most recent async loads are outstanding
// (one K-tile = 6 async b128 per thread).
__device__ __forceinline__ void wait_async_le6() {
  asm volatile("s_wait_asynccnt 0x6" ::: "memory");
}

// ---------------------------------------------------------------------------
// Weight convert + transpose: in f32 [K,N] row-major -> out bf16 [N,K].
// grid.z slices (per-head weights).
// ---------------------------------------------------------------------------
__global__ __launch_bounds__(256) void transpose_f32_bf16(
    const float* __restrict__ in, unsigned short* __restrict__ out,
    int K, int N) {
  size_t slice = (size_t)K * N;
  in  += (size_t)blockIdx.z * slice;
  out += (size_t)blockIdx.z * slice;
  size_t idx = (size_t)blockIdx.x * 256 + threadIdx.x;
  if (idx >= slice) return;
  int n = (int)(idx / K);
  int k = (int)(idx % K);
  out[idx] = f2bf(in[(size_t)k * N + n]);
}

// v [B,T,D] bf16 -> vT [B,D,T] bf16
__global__ __launch_bounds__(256) void transpose_v(
    const unsigned short* __restrict__ in, unsigned short* __restrict__ out) {
  size_t idx = (size_t)blockIdx.x * 256 + threadIdx.x;
  int b = (int)(idx / ((size_t)D_ * T_));
  size_t rem = idx % ((size_t)D_ * T_);
  int d = (int)(rem / T_), t = (int)(rem % T_);
  out[idx] = in[((size_t)b * T_ + t) * D_ + d];
}

// ---------------------------------------------------------------------------
// RMSNorm: one block per row, f32 in -> bf16 out.
// ---------------------------------------------------------------------------
__global__ __launch_bounds__(256) void rmsnorm_kernel(
    const float* __restrict__ x, const float* __restrict__ g,
    unsigned short* __restrict__ out) {
  const int row = blockIdx.x;
  const float* xr = x + (size_t)row * D_;
  float ss = 0.f;
  for (int i = threadIdx.x; i < D_; i += 256) { float v = xr[i]; ss += v * v; }
  for (int m = 1; m < 32; m <<= 1) ss += __shfl_xor(ss, m);
  __shared__ float red[8];
  const int wid = threadIdx.x >> 5, lane = threadIdx.x & 31;
  if (lane == 0) red[wid] = ss;
  __syncthreads();
  float tot = 0.f;
#pragma unroll
  for (int w = 0; w < 8; ++w) tot += red[w];
  const float rs = rsqrtf(tot * (1.0f / D_) + EPS_);
  unsigned short* outr = out + (size_t)row * D_;
  for (int i = threadIdx.x; i < D_; i += 256) outr[i] = f2bf(xr[i] * rs * g[i]);
}

// ---------------------------------------------------------------------------
// GLU: h [M,8D] bf16 -> glu [M,4D] bf16 ; glu = a * sigmoid(b)
// ---------------------------------------------------------------------------
__global__ __launch_bounds__(256) void glu_kernel(
    const unsigned short* __restrict__ h, unsigned short* __restrict__ out) {
  size_t idx = (size_t)blockIdx.x * 256 + threadIdx.x;
  int r = (int)(idx >> 12);            // /4096
  int j = (int)(idx & 4095);
  float a = bf2f(h[(size_t)r * 8192 + j]);
  float b = bf2f(h[(size_t)r * 8192 + 4096 + j]);
  out[idx] = f2bf(a * (1.f / (1.f + __expf(-b))));
}

// ---------------------------------------------------------------------------
// Generic WMMA GEMM: C[M,N] = A[M,K] * Bt[N,K]^T (+bias)(+residual).
// A, Bt bf16 row-major.  Block tile 128x64, 8 waves of 32x32, K-tile 64.
// Double-buffered async global->LDS pipeline: while tile i is consumed by
// WMMAs, tile i+1 streams into the other LDS buffer (ASYNCcnt-tracked).
// EPI 0: bf16 store.  EPI 1: bf16 store + bias.  EPI 2: f32 + bias + residual.
// grid.z = heads: Bt += z*bt_zstride, output col += z*c_col_zoff.
// ---------------------------------------------------------------------------
template <int EPI>
__global__ __launch_bounds__(256) void gemm_bf16(
    const unsigned short* __restrict__ A, int lda,
    const unsigned short* __restrict__ Bt, int ldb, int K,
    void* __restrict__ Cout, int ldc,
    const float* __restrict__ bias, const float* __restrict__ resid,
    long bt_zstride, int c_col_zoff) {
  // rows padded to 72 elems (144B): 16B-aligned b128 frags, conflict-free banks
  __shared__ __align__(16) unsigned short As[2][128 * 72];   // 2 x 18KB
  __shared__ __align__(16) unsigned short Bs[2][64 * 72];    // 2 x  9KB

  const int tid = threadIdx.x;
  const int wid = tid >> 5, lane = tid & 31;
  const int wm = wid & 3, wn = wid >> 2;                 // 4x2 wave grid
  const int laneH = lane & 15;
  const bool hi = lane >= 16;
  const int m0 = blockIdx.x * 128;
  const int n0 = blockIdx.y * 64;
  const unsigned short* Bth = Bt + (size_t)blockIdx.z * (size_t)bt_zstride;

  // per-thread staging assignments (fixed for whole K loop)
  const int ra = tid >> 1, ha = tid & 1;                 // A: 64B each (4xb128)
  const int rb = tid >> 2, qq = tid & 3;                 // Bt: 32B each (2xb128)
  unsigned lofA[2], lofB[2];
  lofA[0] = (unsigned)(uintptr_t)(&As[0][ra * 72 + ha * 32]);
  lofA[1] = (unsigned)(uintptr_t)(&As[1][ra * 72 + ha * 32]);
  lofB[0] = (unsigned)(uintptr_t)(&Bs[0][rb * 72 + qq * 16]);
  lofB[1] = (unsigned)(uintptr_t)(&Bs[1][rb * 72 + qq * 16]);
  const unsigned short* gA = A   + (size_t)(m0 + ra) * lda + ha * 32;
  const unsigned short* gB = Bth + (size_t)(n0 + rb) * ldb + qq * 16;

  auto issue_tile = [&](int buf, int k0) {
    async_load_b128(lofA[buf],      gA + k0);
    async_load_b128(lofA[buf] + 16, gA + k0 + 8);
    async_load_b128(lofA[buf] + 32, gA + k0 + 16);
    async_load_b128(lofA[buf] + 48, gA + k0 + 24);
    async_load_b128(lofB[buf],      gB + k0);
    async_load_b128(lofB[buf] + 16, gB + k0 + 8);
  };

  v8f acc[2][2];
#pragma unroll
  for (int i = 0; i < 2; ++i)
#pragma unroll
    for (int j = 0; j < 2; ++j) acc[i][j] = (v8f){};

  issue_tile(0, 0);                         // prologue: stage tile 0
  int cur = 0;
  for (int k0 = 0; k0 < K; k0 += 64) {
    // stage next tile into the other buffer (re-stage current on last iter
    // so the asynccnt<=6 invariant always covers the tile we consume)
    const int kn = (k0 + 64 < K) ? (k0 + 64) : k0;
    issue_tile(cur ^ 1, kn);
    if (k0 + 128 < K) {                     // prefetch tile after next
      __builtin_prefetch(gA + k0 + 128, 0, 0);
      __builtin_prefetch(gB + k0 + 128, 0, 0);
    }
    wait_async_le6();                       // current tile fully deposited
    __syncthreads();                        // ... by all waves

    const unsigned short* Ab = &As[cur][0];
    const unsigned short* Bb = &Bs[cur][0];
#pragma unroll
    for (int kc = 0; kc < 2; ++kc) {        // two 32-deep WMMA steps per tile
      Frag16 af[2], bfr[2];
#pragma unroll
      for (int sm = 0; sm < 2; ++sm) {
        const unsigned short* p =
            Ab + (wm * 32 + sm * 16 + laneH) * 72 + kc * 32 + (hi ? 16 : 0);
        af[sm].q[0] = *(const uint4*)(p);
        af[sm].q[1] = *(const uint4*)(p + 8);
      }
#pragma unroll
      for (int sn = 0; sn < 2; ++sn) {
        const unsigned short* p =
            Bb + (wn * 32 + sn * 16 + laneH) * 72 + kc * 32 + (hi ? 16 : 0);
        bfr[sn].q[0] = *(const uint4*)(p);
        bfr[sn].q[1] = *(const uint4*)(p + 8);
      }
#pragma unroll
      for (int sm = 0; sm < 2; ++sm)
#pragma unroll
        for (int sn = 0; sn < 2; ++sn)
          acc[sm][sn] = wmma_bf16(af[sm].v, bfr[sn].v, acc[sm][sn]);
    }
    __syncthreads();                        // reads done before buffer refill
    cur ^= 1;
  }

  // --- epilogue (C layout: VGPR r = row r / r+8 for lane groups) ---
  const int zco = blockIdx.z * c_col_zoff;
#pragma unroll
  for (int sm = 0; sm < 2; ++sm)
#pragma unroll
    for (int sn = 0; sn < 2; ++sn)
#pragma unroll
      for (int r = 0; r < 8; ++r) {
        int row = m0 + wm * 32 + sm * 16 + r + (hi ? 8 : 0);
        int gcol = zco + n0 + wn * 32 + sn * 16 + laneH;
        float v = acc[sm][sn][r];
        if (EPI >= 1) v += bias[gcol];
        if (EPI == 2) {
          v += resid[(size_t)row * ldc + gcol];
          ((float*)Cout)[(size_t)row * ldc + gcol] = v;
        } else {
          ((unsigned short*)Cout)[(size_t)row * ldc + gcol] = f2bf(v);
        }
      }
}

// ---------------------------------------------------------------------------
// Wave-level causal flash attention.  Wave = 16 query rows of one (b,h).
// q,k: [B,T,D] bf16 (head at col h*64).  vT: [B,D,T] bf16.  o: [B,T,D] bf16.
// ---------------------------------------------------------------------------
__global__ __launch_bounds__(256) void flash_attn(
    const unsigned short* __restrict__ q, const unsigned short* __restrict__ k,
    const unsigned short* __restrict__ vT, unsigned short* __restrict__ o) {
  __shared__ __align__(16) unsigned short Pbuf[8][16 * 32];  // 1KB per wave

  const int tid = threadIdx.x, wid = tid >> 5, lane = tid & 31;
  const int laneH = lane & 15;
  const bool hi = lane >= 16;
  const int gw = blockIdx.x * 8 + wid;          // 4096 waves total
  const int qb = gw & 127;                      // T/16
  const int h  = (gw >> 7) & 15;
  const int b  = gw >> 11;
  const int q0 = qb * 16;
  const size_t bT = (size_t)b * T_;

  // --- Q A-fragments (16x32 per frag, HD=64 -> 2 frags), held in VGPRs ---
  Frag16 qf[2];
  {
    const unsigned short* qp = q + (bT + q0 + laneH) * D_ + h * 64;
#pragma unroll
    for (int kk = 0; kk < 2; ++kk) {
      const unsigned short* p = qp + kk * 32 + (hi ? 16 : 0);
      qf[kk].q[0] = *(const uint4*)(p);
      qf[kk].q[1] = *(const uint4*)(p + 8);
    }
  }

  v8f Oa[4];
#pragma unroll
  for (int et = 0; et < 4; ++et) Oa[et] = (v8f){};
  float rmax[8], rsum[8];
#pragma unroll
  for (int r = 0; r < 8; ++r) { rmax[r] = -INFINITY; rsum[r] = 0.f; }

  unsigned short* P = &Pbuf[wid][0];

  for (int kb = 0; kb < q0 + 16; kb += 32) {
    // --- S = Q K^T for 32 keys: 2 N-tiles x 2 K-steps = 4 WMMAs ---
    v8f sacc[2];
#pragma unroll
    for (int nt = 0; nt < 2; ++nt) {
      v8f s = (v8f){};
#pragma unroll
      for (int kk = 0; kk < 2; ++kk) {
        const unsigned short* kp =
            k + (bT + kb + nt * 16 + laneH) * D_ + h * 64 + kk * 32 + (hi ? 16 : 0);
        Frag16 u;
        u.q[0] = *(const uint4*)(kp);
        u.q[1] = *(const uint4*)(kp + 8);
        s = wmma_bf16(qf[kk].v, u.v, s);
      }
      sacc[nt] = s;
    }

    // --- scale, causal mask, per-row block max (half-wave reductions) ---
    float bmax[8];
#pragma unroll
    for (int r = 0; r < 8; ++r) {
      const int rowg = q0 + r + (hi ? 8 : 0);
      float mt = -INFINITY;
#pragma unroll
      for (int nt = 0; nt < 2; ++nt) {
        int col = kb + nt * 16 + laneH;
        float v = sacc[nt][r] * 0.125f;            // HD^-0.5
        if (col > rowg) v = -INFINITY;
        sacc[nt][r] = v;
        mt = fmaxf(mt, v);
      }
#pragma unroll
      for (int m = 1; m < 16; m <<= 1) mt = fmaxf(mt, __shfl_xor(mt, m));
      bmax[r] = mt;
    }

    // --- online softmax update ---
#pragma unroll
    for (int r = 0; r < 8; ++r) {
      float mnew = fmaxf(rmax[r], bmax[r]);
      float corr = __expf(rmax[r] - mnew);         // 0 on first block
      rmax[r] = mnew;
      float psum = 0.f;
#pragma unroll
      for (int nt = 0; nt < 2; ++nt) {
        float pv = __expf(sacc[nt][r] - mnew);
        sacc[nt][r] = pv;
        psum += pv;
      }
#pragma unroll
      for (int m = 1; m < 16; m <<= 1) psum += __shfl_xor(psum, m);
      rsum[r] = rsum[r] * corr + psum;
#pragma unroll
      for (int et = 0; et < 4; ++et) Oa[et][r] *= corr;
    }

    // --- P: C-layout regs -> LDS row-major 16x32 -> A-fragment ---
#pragma unroll
    for (int nt = 0; nt < 2; ++nt)
#pragma unroll
      for (int r = 0; r < 8; ++r)
        P[(r + (hi ? 8 : 0)) * 32 + nt * 16 + laneH] = f2bf(sacc[nt][r]);

    Frag16 pf;
    {
      const unsigned short* pp = P + laneH * 32 + (hi ? 16 : 0);
      pf.q[0] = *(const uint4*)(pp);
      pf.q[1] = *(const uint4*)(pp + 8);
    }

    // --- O += P V : 4 e-tiles, K=32 keys each ---
#pragma unroll
    for (int et = 0; et < 4; ++et) {
      const unsigned short* vp =
          vT + ((size_t)b * D_ + h * 64 + et * 16 + laneH) * T_ + kb + (hi ? 16 : 0);
      Frag16 u;
      u.q[0] = *(const uint4*)(vp);
      u.q[1] = *(const uint4*)(vp + 8);
      Oa[et] = wmma_bf16(pf.v, u.v, Oa[et]);
    }
  }

  // --- normalize + store concat-head output [B,T,D] bf16 ---
#pragma unroll
  for (int r = 0; r < 8; ++r) {
    const float inv = 1.f / rsum[r];
#pragma unroll
    for (int et = 0; et < 4; ++et)
      o[(bT + q0 + r + (hi ? 8 : 0)) * D_ + h * 64 + et * 16 + laneH] =
          f2bf(Oa[et][r] * inv);
  }
}

// ---------------------------------------------------------------------------
// Host launcher
// ---------------------------------------------------------------------------
extern "C" void kernel_launch(void* const* d_in, const int* in_sizes, int n_in,
                              void* d_out, int out_size, void* d_ws, size_t ws_size,
                              hipStream_t stream) {
  (void)in_sizes; (void)n_in; (void)out_size; (void)ws_size;
  const float* x      = (const float*)d_in[0];
  const float* wq     = (const float*)d_in[1];
  const float* wk     = (const float*)d_in[2];
  const float* wv     = (const float*)d_in[3];
  const float* w_proj = (const float*)d_in[4];
  const float* b_proj = (const float*)d_in[5];
  const float* w1     = (const float*)d_in[6];
  const float* b1     = (const float*)d_in[7];
  const float* w2     = (const float*)d_in[8];
  const float* b2     = (const float*)d_in[9];
  const float* g1     = (const float*)d_in[10];
  const float* g2     = (const float*)d_in[11];

  char* ws = (char*)d_ws;
  size_t off = 0;
  auto take = [&](size_t bytes) {
    off = (off + 255) & ~(size_t)255;
    char* p = ws + off;
    off += bytes;
    return p;
  };
  const size_t act = (size_t)M_ * D_;        // 4M elements
  unsigned short* wqT   = (unsigned short*)take((size_t)H_ * HD_ * D_ * 2);
  unsigned short* wkT   = (unsigned short*)take((size_t)H_ * HD_ * D_ * 2);
  unsigned short* wvT   = (unsigned short*)take((size_t)H_ * HD_ * D_ * 2);
  unsigned short* wpT   = (unsigned short*)take((size_t)D_ * D_ * 2);
  unsigned short* w1T   = (unsigned short*)take((size_t)8 * D_ * D_ * 2);
  unsigned short* w2T   = (unsigned short*)take((size_t)4 * D_ * D_ * 2);
  unsigned short* xn    = (unsigned short*)take(act * 2);
  unsigned short* qbuf  = (unsigned short*)take(act * 2);
  unsigned short* kbuf  = (unsigned short*)take(act * 2);
  unsigned short* vbuf  = (unsigned short*)take(act * 2);
  unsigned short* vTb   = (unsigned short*)take(act * 2);
  unsigned short* ob    = (unsigned short*)take(act * 2);
  float*          x1    = (float*)take(act * 4);
  unsigned short* xn2   = (unsigned short*)take(act * 2);
  unsigned short* hb    = (unsigned short*)take((size_t)M_ * 8 * D_ * 2);
  unsigned short* glub  = (unsigned short*)take((size_t)M_ * 4 * D_ * 2);

  // 1) weight convert + transpose to [N,K] bf16
  transpose_f32_bf16<<<dim3((D_ * HD_) / 256, 1, H_), 256, 0, stream>>>(wq, wqT, D_, HD_);
  transpose_f32_bf16<<<dim3((D_ * HD_) / 256, 1, H_), 256, 0, stream>>>(wk, wkT, D_, HD_);
  transpose_f32_bf16<<<dim3((D_ * HD_) / 256, 1, H_), 256, 0, stream>>>(wv, wvT, D_, HD_);
  transpose_f32_bf16<<<dim3((D_ * D_) / 256, 1, 1), 256, 0, stream>>>(w_proj, wpT, D_, D_);
  transpose_f32_bf16<<<dim3((D_ * 8 * D_) / 256, 1, 1), 256, 0, stream>>>(w1, w1T, D_, 8 * D_);
  transpose_f32_bf16<<<dim3((4 * D_ * D_) / 256, 1, 1), 256, 0, stream>>>(w2, w2T, 4 * D_, D_);

  // 2) RMSNorm 1
  rmsnorm_kernel<<<M_, 256, 0, stream>>>(x, g1, xn);

  // 3) QKV projections (per-head N=64 tiles via grid.z)
  const dim3 gQKV(M_ / 128, 1, H_);
  gemm_bf16<0><<<gQKV, 256, 0, stream>>>(xn, D_, wqT, D_, D_, qbuf, D_,
                                         nullptr, nullptr, (long)HD_ * D_, HD_);
  gemm_bf16<0><<<gQKV, 256, 0, stream>>>(xn, D_, wkT, D_, D_, kbuf, D_,
                                         nullptr, nullptr, (long)HD_ * D_, HD_);
  gemm_bf16<0><<<gQKV, 256, 0, stream>>>(xn, D_, wvT, D_, D_, vbuf, D_,
                                         nullptr, nullptr, (long)HD_ * D_, HD_);

  // 4) V -> V^T  ([B,T,D] -> [B,D,T])
  transpose_v<<<(int)(act / 256), 256, 0, stream>>>(vbuf, vTb);

  // 5) causal flash attention (4096 waves / 8 per block)
  flash_attn<<<512, 256, 0, stream>>>(qbuf, kbuf, vTb, ob);

  // 6) out projection + bias + residual -> x1 (f32)
  gemm_bf16<2><<<dim3(M_ / 128, D_ / 64, 1), 256, 0, stream>>>(
      ob, D_, wpT, D_, D_, x1, D_, b_proj, x, 0L, 0);

  // 7) RMSNorm 2
  rmsnorm_kernel<<<M_, 256, 0, stream>>>(x1, g2, xn2);

  // 8) FFN up: h = xn2 @ w1 + b1 -> bf16 [M, 8D]
  gemm_bf16<1><<<dim3(M_ / 128, (8 * D_) / 64, 1), 256, 0, stream>>>(
      xn2, D_, w1T, D_, D_, hb, 8 * D_, b1, nullptr, 0L, 0);

  // 9) GLU
  glu_kernel<<<(int)(((size_t)M_ * 4 * D_) / 256), 256, 0, stream>>>(hb, glub);

  // 10) FFN down + bias + residual -> d_out (f32)
  gemm_bf16<2><<<dim3(M_ / 128, D_ / 64, 1), 256, 0, stream>>>(
      glub, 4 * D_, w2T, 4 * D_, 4 * D_, d_out, D_, b2, x1, 0L, 0);
}
